// CausalSelfAttention_18519898980528
// MI455X (gfx1250) — matrix operand order repaired and checked by reading, not compiled, not run on hardware
//
#include <hip/hip_runtime.h>
#include <hip/hip_bf16.h>

// ---------------- types ----------------
typedef __attribute__((ext_vector_type(16))) __bf16 v16bf;
typedef __attribute__((ext_vector_type(8)))  __bf16 v8bf;
typedef __attribute__((ext_vector_type(8)))  float  v8f;
typedef __attribute__((ext_vector_type(4)))  unsigned int v4u;
typedef __attribute__((ext_vector_type(8)))  int v8i;
typedef __attribute__((ext_vector_type(4)))  int v4i;

#define S_LEN   2048
#define NB      2
#define NH      12
#define HD      64
#define DMODEL  768
#define TOKENS  4096   // NB * S_LEN
#define SCALE_F 0.125f // 64^-0.5

__device__ __forceinline__ v8bf ld8(const __bf16* p) { return *(const v8bf*)p; }

__device__ __forceinline__ v16bf mk16(v8bf lo, v8bf hi) {
  v16bf r;
#pragma unroll
  for (int i = 0; i < 8; ++i) { r[i] = lo[i]; r[i + 8] = hi[i]; }
  return r;
}

__device__ __forceinline__ v8f wmma_bf16(v16bf a, v16bf b, v8f c) {
  // D = A(16x32 bf16) * B(32x16 bf16) + C(16x16 f32)
  return __builtin_amdgcn_wmma_f32_16x16x32_bf16(
      /*neg_a=*/false, a, /*neg_b=*/false, b,
      /*c_mod=*/(short)0, c, /*reuse_a=*/false, /*reuse_b=*/false);
}

// Tensor Data Mover: async 2D tile load (rows x 64 bf16 cols) global -> LDS.
// D# group0: count=1, lds_addr, 57-bit global addr, type=2.
// D# group1: data_size=2B, tensor_dim0=tile_dim0=64, tensor_dim1=tile_dim1=32,
//            tensor_dim0_stride = 768 elements. 2D => groups 2/3 disabled (zeros).
__device__ __forceinline__ void tdm_load_tile_32x64(const __bf16* gptr, unsigned lds_off) {
  const unsigned long long ga = (unsigned long long)(const void*)gptr;
  v4u g0;
  g0[0] = 1u;                                            // count=1 (valid user D#)
  g0[1] = lds_off;                                       // lds_addr (bytes)
  g0[2] = (unsigned)(ga & 0xFFFFFFFFu);                  // global_addr[31:0]
  g0[3] = (unsigned)((ga >> 32) & 0x01FFFFFFu) | (2u << 30); // addr[56:32] | type=2
  v8i g1;
  g1[0] = (int)(1u << 16);    // workgroup_mask=0, data_size=1 (2 bytes)
  g1[1] = (int)(64u << 16);   // tensor_dim0[15:0]=64 in bits[31:16]
  g1[2] = (int)(32u << 16);   // tensor_dim0[31:16]=0 | tensor_dim1[15:0]=32
  g1[3] = (int)(64u << 16);   // tensor_dim1[31:16]=0 | tile_dim0=64
  g1[4] = (int)(32u);         // tile_dim1=32 | tile_dim2=0
  g1[5] = (int)(DMODEL);      // tensor_dim0_stride[31:0] = 768
  g1[6] = 0;                  // stride0[47:32]=0 | stride1[15:0]=0
  g1[7] = 0;                  // stride1[47:16]=0
  v4i z4 = {0, 0, 0, 0};
#if __clang_major__ >= 23
  v8i z8 = {0, 0, 0, 0, 0, 0, 0, 0};
  __builtin_amdgcn_tensor_load_to_lds(g0, g1, z4, z4, z8, 0);
#else
  __builtin_amdgcn_tensor_load_to_lds(g0, g1, z4, z4, 0);
#endif
}

// ---------------- conversion kernels ----------------
__global__ void cvt_f32_bf16(const float* __restrict__ in, __bf16* __restrict__ out, int n) {
  int i = blockIdx.x * 256 + threadIdx.x;
  if (i < n) out[i] = (__bf16)in[i];
}

// out[n][k] = (bf16) in[k][n]   (768x768)
__global__ void transpose_w(const float* __restrict__ in, __bf16* __restrict__ out) {
  int i = blockIdx.x * 256 + threadIdx.x;      // flat output index = n*768 + k
  int n = i / DMODEL, k = i - n * DMODEL;
  out[i] = (__bf16)in[k * DMODEL + n];
}

// ---------------- GEMM: one wave computes a 16x64 tile ----------------
// A: [TOKENS][768] bf16 row-major. Wt: [768][768] bf16, Wt[n][k] = W[k][n].
// mode 0: outb row-major bf16 [TOKENS][768]
// mode 1: outb = Vt[b][h][d][s] bf16  (transposed V for attention)
// mode 2: outf row-major f32  [TOKENS][768]
__global__ void gemm_bf16(const __bf16* __restrict__ A, const __bf16* __restrict__ Wt,
                          const float* __restrict__ bias,
                          __bf16* __restrict__ outb, float* __restrict__ outf, int mode) {
  const int lane = threadIdx.x;
  const int half = lane >> 4;
  const int r    = lane & 15;
  const int mt = blockIdx.x, nt = blockIdx.y;
  const int colbase = nt * 64;

  const __bf16* arow = A  + (size_t)(mt * 16 + r) * DMODEL;
  const __bf16* w0   = Wt + (size_t)(colbase + r) * DMODEL + 8 * half;

  v8f acc[4] = {{}, {}, {}, {}};
#pragma unroll 2
  for (int k0 = 0; k0 < DMODEL; k0 += 32) {
    const v16bf a = mk16(ld8(arow + k0 + 8 * half), ld8(arow + k0 + 16 + 8 * half));
#pragma unroll
    for (int t = 0; t < 4; ++t) {
      const __bf16* wr = w0 + (size_t)(t * 16) * DMODEL;
      const v16bf bfrag = mk16(ld8(wr + k0), ld8(wr + k0 + 16));
      acc[t] = wmma_bf16(a, bfrag, acc[t]);
    }
  }

#pragma unroll
  for (int t = 0; t < 4; ++t) {
    const int   col = colbase + t * 16 + r;
    const float bvv = bias[col];
#pragma unroll
    for (int j = 0; j < 8; ++j) {
      const int   orow = mt * 16 + j + 8 * half;
      const float v    = acc[t][j] + bvv;
      if (mode == 0) {
        outb[(size_t)orow * DMODEL + col] = (__bf16)v;
      } else if (mode == 1) {
        const int s = orow & (S_LEN - 1), bi = orow >> 11;
        const int hh = col >> 6, dl = col & 63;
        outb[((size_t)(bi * NH + hh) * HD + dl) * S_LEN + s] = (__bf16)v;
      } else {
        outf[(size_t)orow * DMODEL + col] = v;
      }
    }
  }
}

// ---------------- flash attention (causal) ----------------
// Q,K: [TOKENS][768] bf16 (row = token, col = h*64+d)
// Vt : [b][h][d][s] bf16.  O: [TOKENS][768] bf16.
// One wave handles one (b, h, 16-query tile). K tiles (32x64) are streamed
// into LDS by the Tensor Data Mover, double-buffered one chunk ahead.
__global__ void attn_kernel(const __bf16* __restrict__ Q, const __bf16* __restrict__ K,
                            const __bf16* __restrict__ Vt, __bf16* __restrict__ O) {
  __shared__ __align__(16) __bf16 Klds[2][32 * 64];   // 8 KB double buffer
  __shared__ __align__(16) __bf16 Plds[16 * 32];      // 1 KB P bounce

  const int lane = threadIdx.x;
  const int half = lane >> 4;
  const int r    = lane & 15;
  const int qt = blockIdx.x, h = blockIdx.y, b = blockIdx.z;
  const int q0 = qt * 16;
  const unsigned klds_off = (unsigned)(unsigned long long)(const void*)&Klds[0][0];

  // Q A-fragments (d = 0..31 and 32..63), resident for the whole kv loop
  const size_t qrow = ((size_t)b * S_LEN + q0 + r) * DMODEL + h * HD;
  const v16bf aq0 = mk16(ld8(Q + qrow +      8 * half), ld8(Q + qrow + 16 + 8 * half));
  const v16bf aq1 = mk16(ld8(Q + qrow + 32 + 8 * half), ld8(Q + qrow + 48 + 8 * half));

  v8f acc0 = {}, acc1 = {}, acc2 = {}, acc3 = {};
  float m[8], l[8];
#pragma unroll
  for (int j = 0; j < 8; ++j) { m[j] = -3.0e38f; l[j] = 0.f; }

  const __bf16* Kbase = K + (size_t)b * S_LEN * DMODEL + h * HD;
  const __bf16* Vbase = Vt + ((size_t)(b * NH + h) * HD) * S_LEN;
  const int nchunks = (q0 + 16 + 31) >> 5;  // kv processed in chunks of 32

  // prime the pipeline: TDM chunk 0 into buffer 0
  tdm_load_tile_32x64(Kbase, klds_off);

  for (int c = 0; c < nchunks; ++c) {
    const int kvb = c * 32;
    const int buf = c & 1;

    // issue next chunk's TDM into the other buffer, then wait for this one
    if (c + 1 < nchunks) {
      tdm_load_tile_32x64(Kbase + (size_t)(kvb + 32) * DMODEL, klds_off + (buf ^ 1) * 4096u);
      __builtin_amdgcn_s_wait_tensorcnt(1);
    } else {
      __builtin_amdgcn_s_wait_tensorcnt(0);
    }
    const __bf16* kb = &Klds[buf][0];

    // ---- scores: two 16x16 tiles (kv columns kvb..+15 and kvb+16..+31) ----
    v8f s0 = {}, s1 = {};
    {
      const __bf16* krow = kb + (0 * 16 + r) * 64;
      v16bf bk0 = mk16(*(const v8bf*)(krow + 8 * half),      *(const v8bf*)(krow + 8 * half + 16));
      v16bf bk1 = mk16(*(const v8bf*)(krow + 32 + 8 * half), *(const v8bf*)(krow + 32 + 8 * half + 16));
      s0 = wmma_bf16(aq0, bk0, s0);
      s0 = wmma_bf16(aq1, bk1, s0);
    }
    {
      const __bf16* krow = kb + (1 * 16 + r) * 64;
      v16bf bk0 = mk16(*(const v8bf*)(krow + 8 * half),      *(const v8bf*)(krow + 8 * half + 16));
      v16bf bk1 = mk16(*(const v8bf*)(krow + 32 + 8 * half), *(const v8bf*)(krow + 32 + 8 * half + 16));
      s1 = wmma_bf16(aq0, bk0, s1);
      s1 = wmma_bf16(aq1, bk1, s1);
    }

    // ---- scale + causal mask ----
#pragma unroll
    for (int j = 0; j < 8; ++j) {
      const int qg = q0 + j + 8 * half;
      float v0 = s0[j] * SCALE_F;
      float v1 = s1[j] * SCALE_F;
      if (kvb + r > qg)      v0 = -3.0e38f;
      if (kvb + 16 + r > qg) v1 = -3.0e38f;
      s0[j] = v0; s1[j] = v1;
    }

    // ---- online softmax update + P into LDS (bf16, row-major 16x32) ----
#pragma unroll
    for (int j = 0; j < 8; ++j) {
      float t = fmaxf(s0[j], s1[j]);
      t = fmaxf(t, __shfl_xor(t, 1, 16));
      t = fmaxf(t, __shfl_xor(t, 2, 16));
      t = fmaxf(t, __shfl_xor(t, 4, 16));
      t = fmaxf(t, __shfl_xor(t, 8, 16));
      const float mnew = fmaxf(m[j], t);
      const float sc   = __expf(m[j] - mnew);
      const float p0   = __expf(s0[j] - mnew);
      const float p1   = __expf(s1[j] - mnew);
      float rs = p0 + p1;
      rs += __shfl_xor(rs, 1, 16);
      rs += __shfl_xor(rs, 2, 16);
      rs += __shfl_xor(rs, 4, 16);
      rs += __shfl_xor(rs, 8, 16);
      l[j] = l[j] * sc + rs;
      m[j] = mnew;
      acc0[j] *= sc; acc1[j] *= sc; acc2[j] *= sc; acc3[j] *= sc;
      Plds[(j + 8 * half) * 32 + r]      = (__bf16)p0;
      Plds[(j + 8 * half) * 32 + 16 + r] = (__bf16)p1;
    }
    __syncthreads();

    // ---- P A-fragment from LDS, then PV accumulation (4 d-tiles) ----
    const v16bf ap = mk16(*(const v8bf*)(Plds + r * 32 + 8 * half),
                          *(const v8bf*)(Plds + r * 32 + 16 + 8 * half));
    const __bf16* vb = Vbase + kvb + 8 * half;
    {
      v16bf bv = mk16(ld8(vb + (size_t)(0 * 16 + r) * S_LEN), ld8(vb + (size_t)(0 * 16 + r) * S_LEN + 16));
      acc0 = wmma_bf16(ap, bv, acc0);
    }
    {
      v16bf bv = mk16(ld8(vb + (size_t)(1 * 16 + r) * S_LEN), ld8(vb + (size_t)(1 * 16 + r) * S_LEN + 16));
      acc1 = wmma_bf16(ap, bv, acc1);
    }
    {
      v16bf bv = mk16(ld8(vb + (size_t)(2 * 16 + r) * S_LEN), ld8(vb + (size_t)(2 * 16 + r) * S_LEN + 16));
      acc2 = wmma_bf16(ap, bv, acc2);
    }
    {
      v16bf bv = mk16(ld8(vb + (size_t)(3 * 16 + r) * S_LEN), ld8(vb + (size_t)(3 * 16 + r) * S_LEN + 16));
      acc3 = wmma_bf16(ap, bv, acc3);
    }
    __syncthreads();
  }

  // ---- normalize + store [token][h*64+d] bf16 ----
#pragma unroll
  for (int j = 0; j < 8; ++j) {
    const float inv   = 1.0f / l[j];
    const int   token = b * S_LEN + q0 + j + 8 * half;
    const size_t base = (size_t)token * DMODEL + h * HD + r;
    O[base +  0] = (__bf16)(acc0[j] * inv);
    O[base + 16] = (__bf16)(acc1[j] * inv);
    O[base + 32] = (__bf16)(acc2[j] * inv);
    O[base + 48] = (__bf16)(acc3[j] * inv);
  }
}

// ---------------- launch ----------------
extern "C" void kernel_launch(void* const* d_in, const int* in_sizes, int n_in,
                              void* d_out, int out_size, void* d_ws, size_t ws_size,
                              hipStream_t stream) {
  (void)in_sizes; (void)n_in; (void)out_size; (void)ws_size;
  const float* x  = (const float*)d_in[0];
  const float* Wq = (const float*)d_in[1];
  const float* bq = (const float*)d_in[2];
  const float* Wk = (const float*)d_in[3];
  const float* bk = (const float*)d_in[4];
  const float* Wv = (const float*)d_in[5];
  const float* bv = (const float*)d_in[6];
  const float* Wo = (const float*)d_in[7];
  const float* bo = (const float*)d_in[8];
  float* out = (float*)d_out;

  // workspace carve-up (~36 MB total)
  char* ws = (char*)d_ws;
  size_t off = 0;
  auto take = [&](size_t elems) -> __bf16* {
    __bf16* p = (__bf16*)(ws + off);
    off = (off + elems * sizeof(__bf16) + 255) & ~(size_t)255;
    return p;
  };
  __bf16* xb  = take((size_t)TOKENS * DMODEL);
  __bf16* WqT = take((size_t)DMODEL * DMODEL);
  __bf16* WkT = take((size_t)DMODEL * DMODEL);
  __bf16* WvT = take((size_t)DMODEL * DMODEL);
  __bf16* WoT = take((size_t)DMODEL * DMODEL);
  __bf16* Qb  = take((size_t)TOKENS * DMODEL);
  __bf16* Kb  = take((size_t)TOKENS * DMODEL);
  __bf16* Vtb = take((size_t)TOKENS * DMODEL);
  __bf16* Ob  = take((size_t)TOKENS * DMODEL);

  const int nx = TOKENS * DMODEL;
  cvt_f32_bf16<<<dim3(nx / 256), dim3(256), 0, stream>>>(x, xb, nx);
  const int nw = DMODEL * DMODEL;
  transpose_w<<<dim3(nw / 256), dim3(256), 0, stream>>>(Wq, WqT);
  transpose_w<<<dim3(nw / 256), dim3(256), 0, stream>>>(Wk, WkT);
  transpose_w<<<dim3(nw / 256), dim3(256), 0, stream>>>(Wv, WvT);
  transpose_w<<<dim3(nw / 256), dim3(256), 0, stream>>>(Wo, WoT);

  dim3 ggrid(TOKENS / 16, DMODEL / 64);
  gemm_bf16<<<ggrid, dim3(32), 0, stream>>>(xb, WqT, bq, Qb,  nullptr, 0);
  gemm_bf16<<<ggrid, dim3(32), 0, stream>>>(xb, WkT, bk, Kb,  nullptr, 0);
  gemm_bf16<<<ggrid, dim3(32), 0, stream>>>(xb, WvT, bv, Vtb, nullptr, 1);

  attn_kernel<<<dim3(S_LEN / 16, NH, NB), dim3(32), 0, stream>>>(Qb, Kb, Vtb, Ob);

  gemm_bf16<<<ggrid, dim3(32), 0, stream>>>(Ob, WoT, bo, nullptr, out, 2);
}
